// MultiHeadSelfAttention_60266981097855
// MI455X (gfx1250) — matrix operand order repaired and checked by reading, not compiled, not run on hardware
//
#include <hip/hip_runtime.h>
#include <hip/hip_bf16.h>
#include <math.h>

// ---------------- problem constants ----------------
#define D_MODEL   1024
#define NHEADS    16
#define HEAD_DIM  64
#define BATCH     2
#define SEQ       2048
#define MTOT      (BATCH*SEQ)     // 4096 rows
#define NQKV      (3*D_MODEL)     // 3072
#define LOG2_THETA 13.287712379549449f  // log2(10000)

// ---------------- WMMA types ----------------
typedef __bf16 bf16_t;
typedef bf16_t v16bf __attribute__((ext_vector_type(16)));
typedef bf16_t v8bf  __attribute__((ext_vector_type(8)));
typedef float  v8f   __attribute__((ext_vector_type(8)));

#define WMMA_BF16(a,b,c) \
  __builtin_amdgcn_wmma_f32_16x16x32_bf16(false,(a),false,(b),(short)0,(c),false,false)

// fp32 -> bf16 round-to-nearest-even, done on u16 to avoid __bf16 scalar math
__device__ __host__ inline unsigned short f2bf(float f) {
  union { float f; unsigned int u; } x; x.f = f;
  unsigned int u = x.u;
  u += 0x7FFFu + ((u >> 16) & 1u);
  return (unsigned short)(u >> 16);
}

// ---- fragment loaders (layouts per CDNA5 ISA §7.12.2) ----
union AFragU { v16bf v; v8bf h[2]; };

// A: 16x32 bf16 tile, row-major source with leading dim ldk (elements).
// lane L: row = L&15; chunk0 @ K = (L>>4)*8, chunk1 @ K = (L>>4)*8 + 16.
__device__ inline v16bf load_a_frag(const unsigned short* base, int ldk, int lane) {
  const unsigned short* p = base + (size_t)(lane & 15) * ldk + ((lane >> 4) << 3);
  AFragU f;
  f.h[0] = *(const v8bf*)(p);
  f.h[1] = *(const v8bf*)(p + 16);
  return f.v;
}

// B: 32x16 bf16 tile given as B^T row-major [N, K] with leading dim ldk.
// lane L: column = L&15; chunk0 @ K = (L>>4)*8, chunk1 @ K = (L>>4)*8 + 16.
__device__ inline v16bf load_b_frag(const unsigned short* base, int ldk, int lane) {
  const unsigned short* p = base + (size_t)(lane & 15) * ldk + ((lane >> 4) << 3);
  AFragU f;
  f.h[0] = *(const v8bf*)(p);
  f.h[1] = *(const v8bf*)(p + 16);
  return f.v;
}

// ---------------- kernel 0: fp32 -> bf16 conversion ----------------
__global__ void cvt_f32_bf16_kernel(const float* __restrict__ src,
                                    unsigned short* __restrict__ dst, int n) {
  int i = blockIdx.x * blockDim.x + threadIdx.x;
  int stride = gridDim.x * blockDim.x;
  for (; i < n; i += stride) dst[i] = f2bf(src[i]);
}

// ---------------- kernel 1: QKV GEMM + fused RoPE ----------------
// xb:[4096,1024] bf16, wq:[3072,1024] bf16 (== B^T layout)
// Q:[b,h,t,d] (scaled by 1/sqrt(64)), K:[b,h,t,d], Vt:[b,h,d,t]  (all bf16)
__global__ __launch_bounds__(256)
void qkv_rope_kernel(const unsigned short* __restrict__ xb,
                     const unsigned short* __restrict__ wq,
                     unsigned short* __restrict__ Q,
                     unsigned short* __restrict__ Kc,
                     unsigned short* __restrict__ Vt) {
  const int lane = threadIdx.x & 31;
  const int wave = threadIdx.x >> 5;
  const int nbase = blockIdx.x * 64;               // 48 blocks
  const int mbase = blockIdx.y * 256 + wave * 32;  // 16 blocks * 8 waves

  v8f acc[2][4] = {};
  for (int k0 = 0; k0 < D_MODEL; k0 += 32) {
    v16bf a0 = load_a_frag(xb + (size_t)mbase       * D_MODEL + k0, D_MODEL, lane);
    v16bf a1 = load_a_frag(xb + (size_t)(mbase + 16) * D_MODEL + k0, D_MODEL, lane);
#pragma unroll
    for (int nt = 0; nt < 4; ++nt) {
      v16bf bf = load_b_frag(wq + (size_t)(nbase + nt * 16) * D_MODEL + k0, D_MODEL, lane);
      acc[0][nt] = WMMA_BF16(a0, bf, acc[0][nt]);
      acc[1][nt] = WMMA_BF16(a1, bf, acc[1][nt]);
    }
  }

  const int halfrow = (lane >> 4) * 8;
  const int lcol = lane & 15;
#pragma unroll
  for (int mi = 0; mi < 2; ++mi) {
#pragma unroll
    for (int nt = 0; nt < 4; ++nt) {
      const int col = nbase + nt * 16 + lcol;   // 0..3071
      const int sec = col >> 10;                // 0=Q 1=K 2=V (uniform per tile)
      const int hh  = (col & 1023) >> 6;
      const int d   = col & 63;
#pragma unroll
      for (int v = 0; v < 8; ++v) {
        const int row = mbase + mi * 16 + halfrow + v;   // 0..4095
        const int bb  = row >> 11;
        const int t   = row & (SEQ - 1);
        float c = acc[mi][nt][v];
        float res;
        if (sec < 2) {
          // RoPE: pair (2i, 2i+1) lives in adjacent lanes (col parity = lane parity)
          float ang = (float)t * exp2f(-(float)(d & ~1) * (LOG2_THETA / 64.0f));
          float sn, cs;
          __sincosf(ang, &sn, &cs);
          float other = __shfl_xor(c, 1, 32);
          const bool even = (d & 1) == 0;
          float xe = even ? c : other;
          float xo = even ? other : c;
          res = even ? (xe * cs - xo * sn) : (xe * sn + xo * cs);
        } else {
          res = c;
        }
        const size_t bh = (size_t)bb * NHEADS + hh;
        if (sec == 0)      Q [(bh * SEQ + t) * HEAD_DIM + d] = f2bf(res * 0.125f);
        else if (sec == 1) Kc[(bh * SEQ + t) * HEAD_DIM + d] = f2bf(res);
        else               Vt[(bh * HEAD_DIM + d) * SEQ + t] = f2bf(res);
      }
    }
  }
}

// ---------------- kernel 2: flash attention (causal, online softmax) ----------
// grid.x = SEQ/128, grid.y = B*H; 8 waves, each owns 16 q-rows.
__global__ __launch_bounds__(256)
void flash_attn_kernel(const unsigned short* __restrict__ Q,
                       const unsigned short* __restrict__ Kc,
                       const unsigned short* __restrict__ Vt,
                       unsigned short* __restrict__ AO) {
  __shared__ __align__(16) unsigned short Plds[8][16][72];  // per-wave P tile, 144B rows

  const int lane = threadIdx.x & 31;
  const int wave = threadIdx.x >> 5;
  const int bh   = blockIdx.y;
  const int b    = bh >> 4, h = bh & 15;
  const int qbase = blockIdx.x * 128 + wave * 16;            // token row base
  const size_t headQK = ((size_t)b * NHEADS + h) * SEQ * HEAD_DIM;
  const size_t headV  = ((size_t)b * NHEADS + h) * HEAD_DIM * SEQ;

  // Q fragments live in registers for the whole loop (2 k-steps over d=64)
  v16bf qa[2];
#pragma unroll
  for (int ks = 0; ks < 2; ++ks)
    qa[ks] = load_a_frag(Q + headQK + (size_t)qbase * HEAD_DIM + ks * 32, HEAD_DIM, lane);

  v8f o[4] = {};
  float mrow[8], srow[8];
#pragma unroll
  for (int v = 0; v < 8; ++v) { mrow[v] = -INFINITY; srow[v] = 0.f; }

  const int halfrow = (lane >> 4) * 8;
  const int lcol = lane & 15;
  const int nkb = (qbase + 16 + 63) >> 6;   // causal: only blocks touching rows

  for (int kb = 0; kb < nkb; ++kb) {
    const int kcol0 = kb * 64;

    // S = (Q*scale) @ K^T  (scale folded into Q)
    v8f st[4] = {};
#pragma unroll
    for (int ks = 0; ks < 2; ++ks) {
#pragma unroll
      for (int ct = 0; ct < 4; ++ct) {
        v16bf kf = load_b_frag(Kc + headQK + (size_t)(kcol0 + ct * 16) * HEAD_DIM + ks * 32,
                               HEAD_DIM, lane);
        st[ct] = WMMA_BF16(qa[ks], kf, st[ct]);
      }
    }

    // causal mask + per-row max (butterfly across 16-lane halves)
    float rmax[8];
#pragma unroll
    for (int v = 0; v < 8; ++v) rmax[v] = -3.0e38f;
#pragma unroll
    for (int ct = 0; ct < 4; ++ct) {
      const int col = kcol0 + ct * 16 + lcol;
#pragma unroll
      for (int v = 0; v < 8; ++v) {
        const int row = qbase + halfrow + v;
        float x = st[ct][v];
        if (col > row) x = -3.0e38f;
        st[ct][v] = x;
        rmax[v] = fmaxf(rmax[v], x);
      }
    }
#pragma unroll
    for (int v = 0; v < 8; ++v)
#pragma unroll
      for (int off = 1; off < 16; off <<= 1)
        rmax[v] = fmaxf(rmax[v], __shfl_xor(rmax[v], off, 32));

    float alpha[8], psum[8];
#pragma unroll
    for (int v = 0; v < 8; ++v) {
      float mnew = fmaxf(mrow[v], rmax[v]);
      alpha[v] = __expf(mrow[v] - mnew);   // 0 on first block (exp(-inf))
      mrow[v] = mnew;
      psum[v] = 0.f;
    }

    // P = exp(S - m) -> per-wave LDS tile (relayout C-layout -> A-layout)
#pragma unroll
    for (int ct = 0; ct < 4; ++ct) {
#pragma unroll
      for (int v = 0; v < 8; ++v) {
        float p = __expf(st[ct][v] - mrow[v]);
        psum[v] += p;
        Plds[wave][halfrow + v][ct * 16 + lcol] = f2bf(p);
      }
    }
#pragma unroll
    for (int v = 0; v < 8; ++v) {
#pragma unroll
      for (int off = 1; off < 16; off <<= 1)
        psum[v] += __shfl_xor(psum[v], off, 32);
      srow[v] = srow[v] * alpha[v] + psum[v];
    }
#pragma unroll
    for (int nt = 0; nt < 4; ++nt)
#pragma unroll
      for (int v = 0; v < 8; ++v)
        o[nt][v] *= alpha[v];

    // O += P @ V   (Vt is [d, t] so B^T rows are contiguous)
#pragma unroll
    for (int ks = 0; ks < 2; ++ks) {
      v16bf pa = load_a_frag(&Plds[wave][0][0] + ks * 32, 72, lane);
#pragma unroll
      for (int nt = 0; nt < 4; ++nt) {
        v16bf vf = load_b_frag(Vt + headV + (size_t)(nt * 16) * SEQ + kcol0 + ks * 32,
                               SEQ, lane);
        o[nt] = WMMA_BF16(pa, vf, o[nt]);
      }
    }
  }

  // normalize and store as bf16 [B, T, H*64] for the out-projection GEMM
#pragma unroll
  for (int nt = 0; nt < 4; ++nt) {
#pragma unroll
    for (int v = 0; v < 8; ++v) {
      const int trow = qbase + halfrow + v;
      float val = o[nt][v] / srow[v];
      AO[((size_t)(b * SEQ + trow)) * D_MODEL + h * HEAD_DIM + nt * 16 + lcol] = f2bf(val);
    }
  }
}

// ---------------- kernel 3: output projection -> fp32 ----------------
__global__ __launch_bounds__(256)
void oproj_kernel(const unsigned short* __restrict__ AOb,
                  const unsigned short* __restrict__ wo,
                  float* __restrict__ out) {
  const int lane = threadIdx.x & 31;
  const int wave = threadIdx.x >> 5;
  const int nbase = blockIdx.x * 64;
  const int mbase = blockIdx.y * 256 + wave * 32;

  v8f acc[2][4] = {};
  for (int k0 = 0; k0 < D_MODEL; k0 += 32) {
    v16bf a0 = load_a_frag(AOb + (size_t)mbase       * D_MODEL + k0, D_MODEL, lane);
    v16bf a1 = load_a_frag(AOb + (size_t)(mbase + 16) * D_MODEL + k0, D_MODEL, lane);
#pragma unroll
    for (int nt = 0; nt < 4; ++nt) {
      v16bf bf = load_b_frag(wo + (size_t)(nbase + nt * 16) * D_MODEL + k0, D_MODEL, lane);
      acc[0][nt] = WMMA_BF16(a0, bf, acc[0][nt]);
      acc[1][nt] = WMMA_BF16(a1, bf, acc[1][nt]);
    }
  }
  const int halfrow = (lane >> 4) * 8;
  const int lcol = lane & 15;
#pragma unroll
  for (int mi = 0; mi < 2; ++mi)
#pragma unroll
    for (int nt = 0; nt < 4; ++nt)
#pragma unroll
      for (int v = 0; v < 8; ++v) {
        const int row = mbase + mi * 16 + halfrow + v;
        out[(size_t)row * D_MODEL + nbase + nt * 16 + lcol] = acc[mi][nt][v];
      }
}

// ---------------- launcher ----------------
extern "C" void kernel_launch(void* const* d_in, const int* in_sizes, int n_in,
                              void* d_out, int out_size, void* d_ws, size_t ws_size,
                              hipStream_t stream) {
  const float* x    = (const float*)d_in[0];   // [2,2048,1024]
  const float* Wqkv = (const float*)d_in[1];   // [3072,1024]
  const float* Wo   = (const float*)d_in[2];   // [1024,1024]
  float* out = (float*)d_out;

  char* ws = (char*)d_ws;
  auto take = [&](size_t bytes) {
    char* p = ws;
    ws += (bytes + 255) & ~(size_t)255;
    return p;
  };
  unsigned short* xb    = (unsigned short*)take((size_t)MTOT * D_MODEL * 2);
  unsigned short* wqkvb = (unsigned short*)take((size_t)NQKV * D_MODEL * 2);
  unsigned short* wob   = (unsigned short*)take((size_t)D_MODEL * D_MODEL * 2);
  unsigned short* Qb    = (unsigned short*)take((size_t)BATCH * NHEADS * SEQ * HEAD_DIM * 2);
  unsigned short* Kb    = (unsigned short*)take((size_t)BATCH * NHEADS * SEQ * HEAD_DIM * 2);
  unsigned short* Vtb   = (unsigned short*)take((size_t)BATCH * NHEADS * SEQ * HEAD_DIM * 2);
  unsigned short* AOb   = (unsigned short*)take((size_t)MTOT * D_MODEL * 2);

  // 0) fp32 -> bf16 staging (whole working set then lives in the 192MB L2)
  cvt_f32_bf16_kernel<<<2048, 256, 0, stream>>>(x,    xb,    MTOT * D_MODEL);
  cvt_f32_bf16_kernel<<<2048, 256, 0, stream>>>(Wqkv, wqkvb, NQKV * D_MODEL);
  cvt_f32_bf16_kernel<<<1024, 256, 0, stream>>>(Wo,   wob,   D_MODEL * D_MODEL);

  // 1) QKV projection + fused RoPE (Q pre-scaled by 1/sqrt(64))
  qkv_rope_kernel<<<dim3(NQKV / 64, MTOT / 256), 256, 0, stream>>>(xb, wqkvb, Qb, Kb, Vtb);

  // 2) causal flash attention
  flash_attn_kernel<<<dim3(SEQ / 128, BATCH * NHEADS), 256, 0, stream>>>(Qb, Kb, Vtb, AOb);

  // 3) output projection to fp32
  oproj_kernel<<<dim3(D_MODEL / 64, MTOT / 256), 256, 0, stream>>>(AOb, wob, out);
}
